// MPNN_ENN_Sum_51505247813806
// MI455X (gfx1250) — matrix operand order repaired and checked by reading, not compiled, not run on hardware
//
#include <hip/hip_runtime.h>
#include <hip/hip_bf16.h>

// ---------------------------------------------------------------------------
// MPNN (NNConv edge-network + GRU + Set2Set) for MI455X / gfx1250, wave32.
// Strategy: recompute the per-edge 32x32 matrices A with bf16 WMMA inside the
// message kernel each step (8 x 84 GFLOP bf16 — cheap on CDNA5), keeping the
// whole ~25 MB working set resident in the 192 MB L2 instead of streaming a
// 1.3 GB materialized A from HBM.
// ---------------------------------------------------------------------------

#define Nn   20000
#define Ee   320000
#define Bb   64
#define Dd   32
#define He   128
#define Tt   8
#define S2S  12

typedef __attribute__((ext_vector_type(16))) __bf16 v16bf;
typedef __attribute__((ext_vector_type(8)))  float  v8f;

__device__ __forceinline__ unsigned short f2bf_bits(float f) {
  unsigned u = __float_as_uint(f);
  unsigned r = u + 0x7FFFu + ((u >> 16) & 1u);   // round-to-nearest-even
  return (unsigned short)(r >> 16);
}
__device__ __forceinline__ float bf2f(unsigned short b) {
  return __uint_as_float(((unsigned)b) << 16);
}
__device__ __forceinline__ void atomAddF(float* p, float v) {
  __hip_atomic_fetch_add(p, v, __ATOMIC_RELAXED, __HIP_MEMORY_SCOPE_AGENT);
}
__device__ __forceinline__ void atomMaxF(float* p, float v) {
  if (v >= 0.0f) atomicMax((int*)p, __float_as_int(v));
  else           atomicMin((unsigned int*)p, __float_as_uint(v));
}
__device__ __forceinline__ float sigm(float x) { return 1.0f / (1.0f + expf(-x)); }

// --------------------------------------------------------------------------
// Pre-swizzle We2 (f32 [128,1024]) into bf16 WMMA B-fragment order:
// tile (nt,kb): 32 lanes x 16 bf16 contiguous per lane.
// B 32x16: lanes 0-15 hold K = kb*32+0..15, lanes 16-31 hold K = kb*32+16..31,
// element t within lane -> K offset t; N = nt*16 + (lane&15).
// --------------------------------------------------------------------------
__global__ void k_swizzle_we2(const float* __restrict__ We2,
                              unsigned short* __restrict__ We2s) {
  int idx = blockIdx.x * 256 + threadIdx.x;          // 131072 total
  int t    = idx & 15;
  int lane = (idx >> 4) & 31;
  int kb   = (idx >> 9) & 3;
  int nt   = idx >> 11;
  int k  = kb * 32 + ((lane >= 16) ? 16 : 0) + t;
  int nn = nt * 16 + (lane & 15);
  We2s[idx] = f2bf_bits(We2[k * 1024 + nn]);
}

// Transpose GRU weights [96,32] -> [32,96] for coalesced lane access.
__global__ void k_transpose(const float* __restrict__ Wih,
                            const float* __restrict__ Whh,
                            float* __restrict__ WihT,
                            float* __restrict__ WhhT) {
  int idx = blockIdx.x * 256 + threadIdx.x;          // 6144 total
  if (idx < 3072) {
    int c = idx >> 5, d = idx & 31;
    WihT[d * 96 + c] = Wih[idx];
  } else {
    int i2 = idx - 3072;
    int c = i2 >> 5, d = i2 & 31;
    WhhT[d * 96 + c] = Whh[i2];
  }
}

// h = x @ W_in + b_in   (wave per node, lane = output feature)
__global__ void k_h_init(const float* __restrict__ x,
                         const float* __restrict__ W_in,
                         const float* __restrict__ b_in,
                         float* __restrict__ h) {
  int n = blockIdx.x * 8 + (threadIdx.x >> 5);
  int j = threadIdx.x & 31;
  float xv = (j < 16) ? x[n * 16 + j] : 0.0f;
  float s = 0.0f;
#pragma unroll
  for (int k = 0; k < 16; ++k)
    s += __shfl(xv, k, 32) * W_in[k * 32 + j];
  h[n * 32 + j] = s + b_in[j];
}

// --------------------------------------------------------------------------
// Fused message kernel (per GRU step):
//  phase 1: recompute Hr = relu(edge_data@We1+be1) for a 16-edge tile,
//           stored directly in WMMA A-fragment-swizzled bf16 LDS order.
//  phase 2: A_tile[16,1024] = Hr @ We2 + be2 via v_wmma_f32_16x16x32_bf16
//           (8 waves x 8 n-tiles x 4 k-steps), result bf16 in LDS.
//  phase 3: msg_e = h[src_e] @ A_e, scatter agg[dst_e] += msg via f32 atomics.
// --------------------------------------------------------------------------
__global__ void k_msg(const float* __restrict__ edge_data,
                      const int* __restrict__ edges,
                      const float* __restrict__ h,
                      const float* __restrict__ We1,
                      const float* __restrict__ be1,
                      const unsigned short* __restrict__ We2s,
                      const float* __restrict__ be2,
                      float* __restrict__ agg) {
  __shared__ unsigned short HrS[2048];      // 16x128 bf16, A-fragment order
  __shared__ unsigned short Atile[16384];   // 16x1024 bf16, row-major

  const int tid  = threadIdx.x;
  const int lane = tid & 31;
  const int w    = tid >> 5;
  const int e0   = blockIdx.x * 16;

  // ---- phase 1: layer-1 MLP into swizzled LDS slots ----
#pragma unroll
  for (int i = 0; i < 8; ++i) {
    int s  = tid + 256 * i;
    int t  = s & 15;
    int ln = (s >> 4) & 31;
    int kb = s >> 9;
    int m  = ln & 15;
    int kl = t + ((ln >= 16) ? 16 : 0);
    int k  = kb * 32 + kl;
    const float* ed = edge_data + (long)(e0 + m) * 8;
    float acc = be1[k];
#pragma unroll
    for (int d = 0; d < 8; ++d) acc += ed[d] * We1[d * 128 + k];
    HrS[s] = f2bf_bits(fmaxf(acc, 0.0f));
  }
  __syncthreads();

  // ---- phase 2: WMMA GEMM -> A tile ----
  union ABu { v16bf v; uint4 q[2]; };
  ABu afr[4];
#pragma unroll
  for (int kb = 0; kb < 4; ++kb) {
    const uint4* p = (const uint4*)&HrS[(kb * 32 + lane) * 16];
    afr[kb].q[0] = p[0];
    afr[kb].q[1] = p[1];
  }
  const int ncol  = lane & 15;
  const int rbase = (lane < 16) ? 0 : 8;
  for (int i = 0; i < 8; ++i) {
    int nt = w * 8 + i;
    v8f acc = {};
#pragma unroll
    for (int kb = 0; kb < 4; ++kb) {
      ABu bfr;
      const uint4* bp = (const uint4*)(We2s + ((nt * 4 + kb) * 32 + lane) * 16);
      bfr.q[0] = bp[0];
      bfr.q[1] = bp[1];
      acc = __builtin_amdgcn_wmma_f32_16x16x32_bf16(
          false, afr[kb].v, false, bfr.v, (short)0, acc, false, false);
    }
    int col = nt * 16 + ncol;
    float b2 = be2[col];
#pragma unroll
    for (int r = 0; r < 8; ++r)
      Atile[(rbase + r) * 1024 + col] = f2bf_bits(acc[r] + b2);
  }
  __syncthreads();

  // ---- phase 3: per-edge matvec + scatter (2 edges per wave) ----
#pragma unroll
  for (int i = 0; i < 2; ++i) {
    int row = w * 2 + i;
    int eid = e0 + row;
    int src = edges[eid];
    int dst = edges[Ee + eid];
    float hv = h[src * 32 + lane];
    float acc = 0.0f;
#pragma unroll
    for (int d = 0; d < 32; ++d) {
      float hs = __shfl(hv, d, 32);
      acc += hs * bf2f(Atile[row * 1024 + d * 32 + lane]);
    }
    atomAddF(&agg[dst * 32 + lane], acc);
  }
}

// GRU cell update (wave per node, lane = feature)
__global__ void k_gru(float* __restrict__ h,
                      const float* __restrict__ agg,
                      const float* __restrict__ W_root,
                      const float* __restrict__ b_conv,
                      const float* __restrict__ WihT,
                      const float* __restrict__ WhhT,
                      const float* __restrict__ bih,
                      const float* __restrict__ bhh) {
  int n = blockIdx.x * 8 + (threadIdx.x >> 5);
  int j = threadIdx.x & 31;
  float hj = h[n * 32 + j];
  float aj = agg[n * 32 + j];
  float s = 0.0f;
#pragma unroll
  for (int d = 0; d < 32; ++d)
    s += __shfl(hj, d, 32) * W_root[d * 32 + j];
  float m = fmaxf(s + aj + b_conv[j], 0.0f);

  float gi[3], gh[3];
#pragma unroll
  for (int g = 0; g < 3; ++g) {
    int c = g * 32 + j;
    float a1 = 0.0f, a2 = 0.0f;
#pragma unroll
    for (int d = 0; d < 32; ++d) {
      a1 += __shfl(m, d, 32)  * WihT[d * 96 + c];
      a2 += __shfl(hj, d, 32) * WhhT[d * 96 + c];
    }
    gi[g] = a1 + bih[c];
    gh[g] = a2 + bhh[c];
  }
  float r  = sigm(gi[0] + gh[0]);
  float z  = sigm(gi[1] + gh[1]);
  float nn = tanhf(gi[2] + r * gh[2]);
  h[n * 32 + j] = (1.0f - z) * nn + z * hj;
}

// Set2Set LSTM cell: q_star[B,64], hl=q_cur[B,32] -> q_nxt[B,32], c in place
__global__ void k_lstm(const float* __restrict__ q_star,
                       const float* __restrict__ q_cur,
                       float* __restrict__ c_st,
                       const float* __restrict__ Wih,   // [128,64]
                       const float* __restrict__ Whh,   // [128,32]
                       const float* __restrict__ bih,
                       const float* __restrict__ bhh,
                       float* __restrict__ q_nxt) {
  int idx = blockIdx.x * 256 + threadIdx.x;   // 2048 total
  int b = idx >> 5, j = idx & 31;
  float g4[4];
#pragma unroll
  for (int gg = 0; gg < 4; ++gg) {
    int c = gg * 32 + j;
    float acc = bih[c] + bhh[c];
    for (int k = 0; k < 64; ++k) acc += q_star[b * 64 + k] * Wih[c * 64 + k];
    for (int k = 0; k < 32; ++k) acc += q_cur[b * 32 + k]  * Whh[c * 32 + k];
    g4[gg] = acc;
  }
  float ci = sigm(g4[0]), cf = sigm(g4[1]);
  float cg = tanhf(g4[2]), co = sigm(g4[3]);
  float cn = cf * c_st[b * 32 + j] + ci * cg;
  c_st[b * 32 + j]  = cn;
  q_nxt[b * 32 + j] = co * tanhf(cn);
}

// q_star lo = q, hi = 0; emax = -inf; denom = 0
__global__ void k_att_setup(float* __restrict__ q_star,
                            const float* __restrict__ q,
                            float* __restrict__ emax,
                            float* __restrict__ denom) {
  int idx = blockIdx.x * 256 + threadIdx.x;   // 2048 total
  int b = idx >> 5, j = idx & 31;
  q_star[b * 64 + j]      = q[b * 32 + j];
  q_star[b * 64 + 32 + j] = 0.0f;
  if (j == 0) {
    emax[b]  = -INFINITY;
    denom[b] = 0.0f;
  }
}

// attention logits e[n] = <h[n], q[batch[n]]> + segment max
__global__ void k_att_logits(const float* __restrict__ h,
                             const float* __restrict__ q_star,
                             const int* __restrict__ batch,
                             float* __restrict__ e_log,
                             float* __restrict__ emax) {
  int n = blockIdx.x * 8 + (threadIdx.x >> 5);
  int j = threadIdx.x & 31;
  int b = batch[n];
  float v = h[n * 32 + j] * q_star[b * 64 + j];
#pragma unroll
  for (int off = 16; off > 0; off >>= 1) v += __shfl_down(v, off, 32);
  if (j == 0) {
    e_log[n] = v;
    atomMaxF(&emax[b], v);
  }
}

// w[n] = exp(e - emax[b]); denom[b] += w
__global__ void k_att_weights(const float* __restrict__ e_log,
                              const float* __restrict__ emax,
                              const int* __restrict__ batch,
                              float* __restrict__ w_att,
                              float* __restrict__ denom) {
  int n = blockIdx.x * 256 + threadIdx.x;
  if (n >= Nn) return;
  int b = batch[n];
  float wv = expf(e_log[n] - emax[b]);
  w_att[n] = wv;
  atomAddF(&denom[b], wv);
}

// q_star hi += a_n * h[n]
__global__ void k_att_readout(const float* __restrict__ h,
                              const float* __restrict__ w_att,
                              const float* __restrict__ denom,
                              const int* __restrict__ batch,
                              float* __restrict__ q_star) {
  int n = blockIdx.x * 8 + (threadIdx.x >> 5);
  int j = threadIdx.x & 31;
  int b = batch[n];
  float a = w_att[n] / denom[b];
  atomAddF(&q_star[b * 64 + 32 + j], a * h[n * 32 + j]);
}

// out[b] = q_star[b] @ W_out + b_out
__global__ void k_out(const float* __restrict__ q_star,
                      const float* __restrict__ W_out,
                      const float* __restrict__ b_out,
                      float* __restrict__ out) {
  int b = threadIdx.x;   // 64 threads
  float acc = b_out[0];
  for (int k = 0; k < 64; ++k) acc += q_star[b * 64 + k] * W_out[k];
  out[b] = acc;
}

// ---------------------------------------------------------------------------

extern "C" void kernel_launch(void* const* d_in, const int* in_sizes, int n_in,
                              void* d_out, int out_size, void* d_ws, size_t ws_size,
                              hipStream_t stream) {
  (void)in_sizes; (void)n_in; (void)out_size; (void)ws_size;

  const float* x         = (const float*)d_in[0];
  const float* edge_data = (const float*)d_in[3];
  const int*   edges     = (const int*)d_in[4];
  const int*   batch     = (const int*)d_in[5];
  const float* W_in   = (const float*)d_in[6];
  const float* b_in   = (const float*)d_in[7];
  const float* We1    = (const float*)d_in[8];
  const float* be1    = (const float*)d_in[9];
  const float* We2    = (const float*)d_in[10];
  const float* be2    = (const float*)d_in[11];
  const float* W_root = (const float*)d_in[12];
  const float* b_conv = (const float*)d_in[13];
  const float* gWih   = (const float*)d_in[14];
  const float* gWhh   = (const float*)d_in[15];
  const float* gbih   = (const float*)d_in[16];
  const float* gbhh   = (const float*)d_in[17];
  const float* lWih   = (const float*)d_in[18];
  const float* lWhh   = (const float*)d_in[19];
  const float* lbih   = (const float*)d_in[20];
  const float* lbhh   = (const float*)d_in[21];
  const float* W_out  = (const float*)d_in[22];
  const float* b_out  = (const float*)d_in[23];

  // workspace carve-up (256-byte aligned slabs)
  char* p = (char*)d_ws;
  auto alloc = [&](size_t bytes) -> void* {
    void* r = (void*)p;
    p += (bytes + 255) & ~((size_t)255);
    return r;
  };
  float*          h       = (float*)alloc((size_t)Nn * 32 * 4);
  float*          agg     = (float*)alloc((size_t)Nn * 32 * 4);
  unsigned short* We2s    = (unsigned short*)alloc((size_t)131072 * 2);
  float*          WihT    = (float*)alloc(3072 * 4);
  float*          WhhT    = (float*)alloc(3072 * 4);
  float*          e_log   = (float*)alloc((size_t)Nn * 4);
  float*          w_att   = (float*)alloc((size_t)Nn * 4);
  float*          emax    = (float*)alloc(Bb * 4);
  float*          denom   = (float*)alloc(Bb * 4);
  float*          q_star  = (float*)alloc(Bb * 64 * 4);
  float*          q_a     = (float*)alloc(Bb * 32 * 4);
  float*          q_b     = (float*)alloc(Bb * 32 * 4);
  float*          c_st    = (float*)alloc(Bb * 32 * 4);

  // one-time (per call) preprocessing
  k_swizzle_we2<<<512, 256, 0, stream>>>(We2, We2s);
  k_transpose<<<24, 256, 0, stream>>>(gWih, gWhh, WihT, WhhT);
  k_h_init<<<Nn / 8, 256, 0, stream>>>(x, W_in, b_in, h);

  // T = 8 GRU message-passing steps (T is a fixed scalar input)
  for (int t = 0; t < Tt; ++t) {
    hipMemsetAsync(agg, 0, (size_t)Nn * 32 * 4, stream);
    k_msg<<<Ee / 16, 256, 0, stream>>>(edge_data, edges, h, We1, be1,
                                       We2s, be2, agg);
    k_gru<<<Nn / 8, 256, 0, stream>>>(h, agg, W_root, b_conv,
                                      WihT, WhhT, gbih, gbhh);
  }

  // Set2Set init: q_star = 0, q = 0, c = 0
  hipMemsetAsync(q_star, 0, Bb * 64 * 4, stream);
  hipMemsetAsync(q_a,    0, Bb * 32 * 4, stream);
  hipMemsetAsync(q_b,    0, Bb * 32 * 4, stream);
  hipMemsetAsync(c_st,   0, Bb * 32 * 4, stream);

  for (int s = 0; s < S2S; ++s) {
    float* qc = (s & 1) ? q_b : q_a;
    float* qn = (s & 1) ? q_a : q_b;
    k_lstm<<<8, 256, 0, stream>>>(q_star, qc, c_st, lWih, lWhh, lbih, lbhh, qn);
    k_att_setup<<<8, 256, 0, stream>>>(q_star, qn, emax, denom);
    k_att_logits<<<Nn / 8, 256, 0, stream>>>(h, q_star, batch, e_log, emax);
    k_att_weights<<<(Nn + 255) / 256, 256, 0, stream>>>(e_log, emax, batch,
                                                        w_att, denom);
    k_att_readout<<<Nn / 8, 256, 0, stream>>>(h, w_att, denom, batch, q_star);
  }

  k_out<<<1, 64, 0, stream>>>(q_star, W_out, b_out, (float*)d_out);
}
